// TransformerBlock_39230231281736
// MI455X (gfx1250) — compile-verified, matrix operand order checked by reading, not run on hardware
//
#include <hip/hip_runtime.h>
#include <hip/hip_bf16.h>
#include <math.h>

#define NNODES 40000
#define CDIM   128
#define HHEADS 8
#define DHEAD  16
#define KNEI   8
#define NKE    (NNODES * KNEI)          // 320000 spatial edges
#define NEDGE  (2 * NKE + NNODES)       // 680000 total edges
#define HSIZE  (1u << 21)               // hash slots (>3x edges)
#define HMASK  (HSIZE - 1u)
#define EMPTYK 0xFFFFFFFFu

typedef __attribute__((ext_vector_type(16))) _Float16 v16h;
typedef __attribute__((ext_vector_type(8)))  _Float16 v8h;
typedef __attribute__((ext_vector_type(4)))  _Float16 v4h;
typedef __attribute__((ext_vector_type(8)))  float    v8f;

// ---------- helpers ----------

__device__ __forceinline__ unsigned hashk(unsigned k) {
  k *= 2654435761u;
  k ^= k >> 15;
  return k & HMASK;
}

// order-preserving float<->uint encode for atomicMax on floats
__device__ __forceinline__ unsigned encf(float f) {
  unsigned u = __float_as_uint(f);
  return (u & 0x80000000u) ? ~u : (u | 0x80000000u);
}
__device__ __forceinline__ float decf(unsigned u) {
  return __uint_as_float((u & 0x80000000u) ? (u & 0x7FFFFFFFu) : ~u);
}

// edge e -> (i0, i1) per reference concat([spatial, spatial[::-1], temporal])
__device__ __forceinline__ void get_edge(int e, const int* sp, const int* te,
                                         int& i0, int& i1) {
  if (e < NKE)            { i0 = sp[e];        i1 = sp[NKE + e]; }
  else if (e < 2 * NKE)   { int j = e - NKE;  i0 = sp[NKE + j]; i1 = sp[j]; }
  else                    { int j = e - 2*NKE; i0 = te[j];       i1 = te[NNODES + j]; }
}

// Load a 16x32 f16 WMMA fragment (A layout; B loaded from W^T identically).
// Lane L: row = r0 + L%16; K halves: kb..kb+7 and kb+16..kb+23, kb = k0 + 8*(L>=16)
__device__ __forceinline__ v16h load_frag(const _Float16* base, int ld,
                                          int r0, int k0, int lane) {
  int row = r0 + (lane & 15);
  int kb  = k0 + ((lane & 16) ? 8 : 0);
  const _Float16* p = base + (size_t)row * ld + kb;
  v8h lo = *(const v8h*)p;
  v8h hi = *(const v8h*)(p + 16);
  v16h f;
#pragma unroll
  for (int i = 0; i < 8; ++i) { f[i] = lo[i]; f[i + 8] = hi[i]; }
  return f;
}

__device__ __forceinline__ v8f wmma_f16(v16h a, v16h b, v8f c) {
  return __builtin_amdgcn_wmma_f32_16x16x32_f16(false, a, false, b,
                                                (short)0, c, false, false);
}

// ---------- prep kernels ----------

// W (KxN f32, row-major) -> WT (NxK f16, row-major)
__global__ void transpose_f16(const float* __restrict__ W, _Float16* __restrict__ WT,
                              int K, int N) {
  int idx = blockIdx.x * blockDim.x + threadIdx.x;
  if (idx >= K * N) return;
  int nc = idx / K, kc = idx % K;
  WT[idx] = (_Float16)W[(size_t)kc * N + nc];
}

__global__ void cvt_f16(const float* __restrict__ in, _Float16* __restrict__ out, int n) {
  int i = blockIdx.x * blockDim.x + threadIdx.x;
  if (i < n) out[i] = (_Float16)in[i];
}

// LayerNorm over C=128, one wave per row, f16 output
__global__ void ln_rows(const float* __restrict__ x, const float* __restrict__ w,
                        const float* __restrict__ b, _Float16* __restrict__ out,
                        int nrows) {
  int wid  = (blockIdx.x * blockDim.x + threadIdx.x) >> 5;
  int lane = threadIdx.x & 31;
  if (wid >= nrows) return;
  const float* row = x + (size_t)wid * CDIM;
  float4 v = ((const float4*)row)[lane];
  float s  = v.x + v.y + v.z + v.w;
  float sq = v.x * v.x + v.y * v.y + v.z * v.z + v.w * v.w;
#pragma unroll
  for (int o = 16; o; o >>= 1) { s += __shfl_xor(s, o, 32); sq += __shfl_xor(sq, o, 32); }
  float mu  = s * (1.f / CDIM);
  float var = sq * (1.f / CDIM) - mu * mu;
  float rs  = rsqrtf(var + 1e-5f);
  int c = lane * 4;
  v4h o4;
  o4[0] = (_Float16)((v.x - mu) * rs * w[c + 0] + b[c + 0]);
  o4[1] = (_Float16)((v.y - mu) * rs * w[c + 1] + b[c + 1]);
  o4[2] = (_Float16)((v.z - mu) * rs * w[c + 2] + b[c + 2]);
  o4[3] = (_Float16)((v.w - mu) * rs * w[c + 3] + b[c + 3]);
  *((v4h*)(out + (size_t)wid * CDIM + c)) = o4;
}

// ---------- WMMA GEMM: D = A(MxK f16) * B^T-loaded(NoutxK f16) + bias ----------
// mode 0: QKV split -> q*scale, k, v (f16, NxC each)
// mode 1: outF = res + val + bias (f32)
// mode 2: outH = gelu(val + bias)  (f16, ld = Nout)

__device__ __forceinline__ void epilogue(v8f acc, int m0, int n0, int lane, int Nout,
                                         int mode, float scale,
                                         const float* __restrict__ bias,
                                         const float* __restrict__ res,
                                         float* __restrict__ outF,
                                         _Float16* __restrict__ oq,
                                         _Float16* __restrict__ ok,
                                         _Float16* __restrict__ ov) {
  int ncol = n0 + (lane & 15);
  float bv = bias[ncol];
  int mb = m0 + ((lane & 16) ? 8 : 0);
#pragma unroll
  for (int r = 0; r < 8; ++r) {
    int m = mb + r;
    float val = acc[r] + bv;
    if (mode == 0) {
      if (ncol < CDIM)          oq[(size_t)m * CDIM + ncol]              = (_Float16)(val * scale);
      else if (ncol < 2 * CDIM) ok[(size_t)m * CDIM + (ncol - CDIM)]     = (_Float16)val;
      else                      ov[(size_t)m * CDIM + (ncol - 2 * CDIM)] = (_Float16)val;
    } else if (mode == 1) {
      outF[(size_t)m * Nout + ncol] = res[(size_t)m * Nout + ncol] + val;
    } else {
      float g = 0.5f * val * (1.0f + erff(val * 0.70710678118f));
      oq[(size_t)m * Nout + ncol] = (_Float16)g;
    }
  }
}

// One wave owns a 16-row stripe; processes two 16-col tiles at a time
// (independent WMMA chains). For K==128 the 4 A fragments stay in registers.
__global__ void gemm16(const _Float16* __restrict__ A, const _Float16* __restrict__ BT,
                       const float* __restrict__ bias, int M, int K, int Nout,
                       int mode, float scale, const float* __restrict__ res,
                       float* __restrict__ outF, _Float16* __restrict__ oq,
                       _Float16* __restrict__ ok, _Float16* __restrict__ ov) {
  int wid  = (blockIdx.x * blockDim.x + threadIdx.x) >> 5;
  int lane = threadIdx.x & 31;
  int nM = M >> 4;
  if (wid >= nM) return;
  int m0 = wid << 4;

  if (K == 128) {
    // cache all A fragments for this row stripe in registers
    v16h a0 = load_frag(A, 128, m0,  0, lane);
    v16h a1 = load_frag(A, 128, m0, 32, lane);
    v16h a2 = load_frag(A, 128, m0, 64, lane);
    v16h a3 = load_frag(A, 128, m0, 96, lane);
    for (int n0 = 0; n0 < Nout; n0 += 32) {
      v8f acc0 = {}, acc1 = {};
      {
        v16h b0 = load_frag(BT, 128, n0,      0, lane);
        v16h b1 = load_frag(BT, 128, n0 + 16, 0, lane);
        acc0 = wmma_f16(a0, b0, acc0);
        acc1 = wmma_f16(a0, b1, acc1);
      }
      {
        v16h b0 = load_frag(BT, 128, n0,      32, lane);
        v16h b1 = load_frag(BT, 128, n0 + 16, 32, lane);
        acc0 = wmma_f16(a1, b0, acc0);
        acc1 = wmma_f16(a1, b1, acc1);
      }
      {
        v16h b0 = load_frag(BT, 128, n0,      64, lane);
        v16h b1 = load_frag(BT, 128, n0 + 16, 64, lane);
        acc0 = wmma_f16(a2, b0, acc0);
        acc1 = wmma_f16(a2, b1, acc1);
      }
      {
        v16h b0 = load_frag(BT, 128, n0,      96, lane);
        v16h b1 = load_frag(BT, 128, n0 + 16, 96, lane);
        acc0 = wmma_f16(a3, b0, acc0);
        acc1 = wmma_f16(a3, b1, acc1);
      }
      epilogue(acc0, m0, n0,      lane, Nout, mode, scale, bias, res, outF, oq, ok, ov);
      epilogue(acc1, m0, n0 + 16, lane, Nout, mode, scale, bias, res, outF, oq, ok, ov);
    }
  } else {
    for (int n0 = 0; n0 < Nout; n0 += 32) {
      v8f acc0 = {}, acc1 = {};
      for (int k0 = 0; k0 < K; k0 += 32) {
        v16h a  = load_frag(A,  K, m0,      k0, lane);
        v16h b0 = load_frag(BT, K, n0,      k0, lane);
        v16h b1 = load_frag(BT, K, n0 + 16, k0, lane);
        acc0 = wmma_f16(a, b0, acc0);
        acc1 = wmma_f16(a, b1, acc1);
      }
      epilogue(acc0, m0, n0,      lane, Nout, mode, scale, bias, res, outF, oq, ok, ov);
      epilogue(acc1, m0, n0 + 16, lane, Nout, mode, scale, bias, res, outF, oq, ok, ov);
    }
  }
}

// ---------- edge-attention kernels ----------

__global__ void edge_insert(const int* __restrict__ sp, const int* __restrict__ te,
                            unsigned* __restrict__ hkeys, unsigned* __restrict__ hmin) {
  int e = blockIdx.x * blockDim.x + threadIdx.x;
  if (e >= NEDGE) return;
  int i0, i1; get_edge(e, sp, te, i0, i1);
  unsigned key = (unsigned)i0 * (unsigned)NNODES + (unsigned)i1;
  unsigned s = hashk(key);
  for (unsigned it = 0; it < HSIZE; ++it) {
    unsigned prev = atomicCAS(&hkeys[s], EMPTYK, key);
    if (prev == EMPTYK || prev == key) { atomicMin(&hmin[s], (unsigned)e); break; }
    s = (s + 1) & HMASK;
  }
}

__global__ void edge_logits(const int* __restrict__ sp, const int* __restrict__ te,
                            const _Float16* __restrict__ qh, const _Float16* __restrict__ kh,
                            const unsigned* __restrict__ hkeys, const unsigned* __restrict__ hmin,
                            float* __restrict__ ebuf, unsigned* __restrict__ segmax) {
  int t = blockIdx.x * blockDim.x + threadIdx.x;
  if (t >= NEDGE * HHEADS) return;
  int e = t >> 3, h = t & 7;
  int i0, i1; get_edge(e, sp, te, i0, i1);
  unsigned key = (unsigned)i0 * (unsigned)NNODES + (unsigned)i1;
  unsigned s = hashk(key);
  unsigned mn = (unsigned)e;
  for (unsigned it = 0; it < HSIZE; ++it) {
    unsigned kk = hkeys[s];
    if (kk == key)    { mn = hmin[s]; break; }
    if (kk == EMPTYK) { break; }
    s = (s + 1) & HMASK;
  }
  float l;
  if (mn != (unsigned)e) {
    l = -1e30f;  // duplicate edge (stable-sort "keep first" semantics)
  } else {
    const _Float16* qp = qh + (size_t)i0 * CDIM + h * DHEAD;
    const _Float16* kp = kh + (size_t)i1 * CDIM + h * DHEAD;
    v8h q0 = *(const v8h*)qp, q1 = *(const v8h*)(qp + 8);
    v8h k0 = *(const v8h*)kp, k1 = *(const v8h*)(kp + 8);
    float acc = 0.f;
#pragma unroll
    for (int i = 0; i < 8; ++i)
      acc += (float)q0[i] * (float)k0[i] + (float)q1[i] * (float)k1[i];
    l = acc;
  }
  ebuf[t] = l;
  atomicMax(&segmax[i0 * HHEADS + h], encf(l));
}

__global__ void edge_exp(const int* __restrict__ sp, const int* __restrict__ te,
                         const unsigned* __restrict__ segmax,
                         float* __restrict__ ebuf, float* __restrict__ denom) {
  int t = blockIdx.x * blockDim.x + threadIdx.x;
  if (t >= NEDGE * HHEADS) return;
  int e = t >> 3, h = t & 7;
  int i0, i1; get_edge(e, sp, te, i0, i1);
  float m  = decf(segmax[i0 * HHEADS + h]);
  float ex = expf(ebuf[t] - m);
  ebuf[t] = ex;
  atomicAdd(&denom[i0 * HHEADS + h], ex);
}

__global__ void edge_accum(const int* __restrict__ sp, const int* __restrict__ te,
                           const float* __restrict__ denom, const float* __restrict__ ebuf,
                           const _Float16* __restrict__ vh, float* __restrict__ xb) {
  int t = blockIdx.x * blockDim.x + threadIdx.x;
  if (t >= NEDGE * HHEADS) return;
  int e = t >> 3, h = t & 7;
  int i0, i1; get_edge(e, sp, te, i0, i1);
  float attn = ebuf[t] / denom[i0 * HHEADS + h];
  const _Float16* vp = vh + (size_t)i1 * CDIM + h * DHEAD;
  float* xp = xb + (size_t)i0 * CDIM + h * DHEAD;
  v8h v0 = *(const v8h*)vp, v1 = *(const v8h*)(vp + 8);
#pragma unroll
  for (int i = 0; i < 8; ++i) {
    atomicAdd(xp + i,     attn * (float)v0[i]);
    atomicAdd(xp + 8 + i, attn * (float)v1[i]);
  }
}

// ---------- launch ----------

extern "C" void kernel_launch(void* const* d_in, const int* in_sizes, int n_in,
                              void* d_out, int out_size, void* d_ws, size_t ws_size,
                              hipStream_t stream) {
  const float* feats = (const float*)d_in[0];
  const int*   te    = (const int*)d_in[2];
  const int*   sp    = (const int*)d_in[3];
  const float* n1w   = (const float*)d_in[5];
  const float* n1b   = (const float*)d_in[6];
  const float* qkvw  = (const float*)d_in[7];
  const float* qkvb  = (const float*)d_in[8];
  const float* pw    = (const float*)d_in[9];
  const float* pb    = (const float*)d_in[10];
  const float* n2w   = (const float*)d_in[11];
  const float* n2b   = (const float*)d_in[12];
  const float* f1w   = (const float*)d_in[13];
  const float* f1b   = (const float*)d_in[14];
  const float* f2w   = (const float*)d_in[15];
  const float* f2b   = (const float*)d_in[16];
  float* out = (float*)d_out;

  char* base = (char*)d_ws;
  size_t off = 0;
  auto alloc = [&](size_t bytes) -> void* {
    void* p = base + off;
    off += (bytes + 255) & ~(size_t)255;
    return p;
  };

  _Float16* h1  = (_Float16*)alloc((size_t)NNODES * CDIM * 2);
  _Float16* qh  = (_Float16*)alloc((size_t)NNODES * CDIM * 2);
  _Float16* kh  = (_Float16*)alloc((size_t)NNODES * CDIM * 2);
  _Float16* vh  = (_Float16*)alloc((size_t)NNODES * CDIM * 2);
  _Float16* h2  = (_Float16*)alloc((size_t)NNODES * CDIM * 2);
  _Float16* mb  = (_Float16*)alloc((size_t)NNODES * 512 * 2);
  float*    f2  = (float*)alloc((size_t)NNODES * CDIM * 4);
  float*    xb  = (float*)alloc((size_t)NNODES * CDIM * 4);
  _Float16* wq  = (_Float16*)alloc((size_t)384 * 128 * 2);
  _Float16* wp  = (_Float16*)alloc((size_t)128 * 128 * 2);
  _Float16* w1  = (_Float16*)alloc((size_t)512 * 128 * 2);
  _Float16* w2  = (_Float16*)alloc((size_t)128 * 512 * 2);
  float*    eb  = (float*)alloc((size_t)NEDGE * HHEADS * 4);
  unsigned* sm  = (unsigned*)alloc((size_t)NNODES * HHEADS * 4);
  float*    dn  = (float*)alloc((size_t)NNODES * HHEADS * 4);
  unsigned* hk  = (unsigned*)alloc((size_t)HSIZE * 4);
  unsigned* hm  = (unsigned*)alloc((size_t)HSIZE * 4);

  // init (graph-capturable async memsets)
  hipMemsetAsync(hk, 0xFF, (size_t)HSIZE * 4, stream);
  hipMemsetAsync(hm, 0xFF, (size_t)HSIZE * 4, stream);
  hipMemsetAsync(sm, 0x00, (size_t)NNODES * HHEADS * 4, stream);
  hipMemsetAsync(dn, 0x00, (size_t)NNODES * HHEADS * 4, stream);
  hipMemsetAsync(xb, 0x00, (size_t)NNODES * CDIM * 4, stream);

  // transpose weights to f16 (tiny)
  transpose_f16<<<(128 * 384 + 255) / 256, 256, 0, stream>>>(qkvw, wq, 128, 384);
  transpose_f16<<<(128 * 128 + 255) / 256, 256, 0, stream>>>(pw,   wp, 128, 128);
  transpose_f16<<<(128 * 512 + 255) / 256, 256, 0, stream>>>(f1w,  w1, 128, 512);
  transpose_f16<<<(512 * 128 + 255) / 256, 256, 0, stream>>>(f2w,  w2, 512, 128);

  const int LN_GRID   = NNODES / 8;               // 8 waves/block, 1 row/wave
  const int GEMM_GRID = (NNODES / 16 + 7) / 8;    // 2500 m-tiles, 8 waves/block
  const int E_GRID    = (NEDGE + 255) / 256;
  const int EH_GRID   = (NEDGE * HHEADS + 255) / 256;

  // 1) h1 = LN(feats)
  ln_rows<<<LN_GRID, 256, 0, stream>>>(feats, n1w, n1b, h1, NNODES);

  // 2) qkv = h1 @ qkv_w + b -> q*scale, k, v (f16)
  gemm16<<<GEMM_GRID, 256, 0, stream>>>(h1, wq, qkvb, NNODES, 128, 384,
                                        0, 0.25f, nullptr, nullptr, qh, kh, vh);

  // 3) edge attention (hash dedup, segment softmax via L2 atomics)
  edge_insert<<<E_GRID, 256, 0, stream>>>(sp, te, hk, hm);
  edge_logits<<<EH_GRID, 256, 0, stream>>>(sp, te, qh, kh, hk, hm, eb, sm);
  edge_exp   <<<EH_GRID, 256, 0, stream>>>(sp, te, sm, eb, dn);
  edge_accum <<<EH_GRID, 256, 0, stream>>>(sp, te, dn, eb, vh, xb);

  // 4) x (f32) -> f16 (reuse h1, dead after step 2)
  _Float16* xh = h1;
  cvt_f16<<<(NNODES * CDIM + 255) / 256, 256, 0, stream>>>(xb, xh, NNODES * CDIM);

  // 5) feats2 = feats + x @ proj_w + proj_b
  gemm16<<<GEMM_GRID, 256, 0, stream>>>(xh, wp, pb, NNODES, 128, 128,
                                        1, 1.f, feats, f2, nullptr, nullptr, nullptr);

  // 6) h2 = LN(feats2)
  ln_rows<<<LN_GRID, 256, 0, stream>>>(f2, n2w, n2b, h2, NNODES);

  // 7) m = gelu(h2 @ fc1_w + b) (f16)
  gemm16<<<GEMM_GRID, 256, 0, stream>>>(h2, w1, f1b, NNODES, 128, 512,
                                        2, 1.f, nullptr, nullptr, mb, nullptr, nullptr);

  // 8) out = feats2 + m @ fc2_w + b
  gemm16<<<GEMM_GRID, 256, 0, stream>>>(mb, w2, f2b, NNODES, 512, 128,
                                        1, 1.f, f2, out, nullptr, nullptr, nullptr);
}
